// CasualMultiheadSelfAttention_65231963292188
// MI455X (gfx1250) — matrix operand order repaired and checked by reading, not compiled, not run on hardware
//
#include <hip/hip_runtime.h>

typedef __attribute__((ext_vector_type(16))) _Float16 v16h;
typedef __attribute__((ext_vector_type(8)))  float    v8f;

#define S_LEN   4096
#define DMODEL  768
#define NHEADS  12
#define DK      64
#define TILE    64
#define KC      32

#if __has_builtin(__builtin_amdgcn_tensor_load_to_lds) && __has_builtin(__builtin_amdgcn_s_wait_tensorcnt)
#define HAVE_TDM 1
#else
#define HAVE_TDM 0
#endif

// 16-bit A-matrix 16x32 layout: lane 0-15 = row M, K = {0..15};
// lanes 16-31 use K = {16..31} (contiguous, matching the B fills).
__device__ __forceinline__ int a_kmap(int h, int lane) {
  return h + ((lane >> 4) & 1) * 16;
}

#if HAVE_TDM
// Tensor Data Mover: load a 64x64 f16 tile (row stride 64 elems in memory)
// into LDS with a 4-DWORD pad after every 32 DWORDs => LDS row stride 72
// halves, matching the [TILE][DK+8] arrays. D# layout per
// CDNA5 ISA (group0: count/lds/global/type,
// group1: data_size, pad, tensor dims 64 x 4096, tile 64x64, stride 64).
__device__ __forceinline__ void tdm_load_tile(unsigned int lds_byte_addr,
                                              const _Float16* gsrc) {
  typedef __attribute__((ext_vector_type(4))) unsigned int u32x4;
  typedef __attribute__((ext_vector_type(8))) int          i32x8;
  typedef __attribute__((ext_vector_type(4))) int          i32x4;
  unsigned long long ga = (unsigned long long)(size_t)gsrc;

  u32x4 g0;
  g0[0] = 1u;                                    // count=1, user descriptor
  g0[1] = lds_byte_addr;                         // lds_addr
  g0[2] = (unsigned int)ga;                      // global_addr[31:0]
  g0[3] = (unsigned int)((ga >> 32) & 0x1FFFFFFu) | (2u << 30); // [56:32], type=2

  i32x8 g1;
  g1[0] = (1 << 16)                              // data_size = 2 bytes
        | (1 << 20)                              // pad_enable
        | (4 << 22)                              // pad_interval: 32 DWORDs
        | (3 << 25);                             // pad_amount: 4 DWORDs
  g1[1] = (int)(64u << 16);                      // tensor_dim0 = 64 (bits 79:48)
  g1[2] = (int)(4096u << 16);                    // tensor_dim1 = 4096 (bits 111:80)
  g1[3] = (int)(64u << 16);                      // tile_dim0 = 64 (bits 127:112)
  g1[4] = 64;                                    // tile_dim1 = 64, tile_dim2 = 0
  g1[5] = 64;                                    // tensor_dim0_stride = 64
  g1[6] = 0;
  g1[7] = 0;

  i32x4 z4 = {0, 0, 0, 0};
#if __clang_major__ >= 23
  i32x8 z8 = {0, 0, 0, 0, 0, 0, 0, 0};
  __builtin_amdgcn_tensor_load_to_lds(g0, g1, z4, z4, z8, 0);
#else
  __builtin_amdgcn_tensor_load_to_lds(g0, g1, z4, z4, 0);
#endif
}
#endif

// ---------------------------------------------------------------------------
// Pass 1: Q/K/V projection (x @ w.T) with fused RoPE, f16 head-major output.
// grid = (S/64, heads, 3), block = 256 (8 waves, 2 16x16 subtiles per wave)
// ---------------------------------------------------------------------------
__global__ __launch_bounds__(256)
void qkv_rope_kernel(const float* __restrict__ x,
                     const float* __restrict__ wq,
                     const float* __restrict__ wk,
                     const float* __restrict__ wv,
                     const int*   __restrict__ pos,
                     _Float16* __restrict__ qh,
                     _Float16* __restrict__ kh,
                     _Float16* __restrict__ vh)
{
  const int s0    = blockIdx.x * TILE;
  const int head  = blockIdx.y;
  const int n0    = head * TILE;
  const int which = blockIdx.z;
  const float* w  = (which == 0) ? wq : (which == 1) ? wk : wv;
  _Float16*  dst  = (which == 0) ? qh : (which == 1) ? kh : vh;

  __shared__ _Float16 At[TILE][KC + 8];
  __shared__ _Float16 Wt[TILE][KC + 8];

  const int tid   = threadIdx.x;
  const int wave  = tid >> 5;
  const int lane  = tid & 31;
  const int tm    = wave >> 1;
  const int tn0   = (wave & 1) * 2;
  const int mbase = (lane >> 4) * 8;
  const int nlane = lane & 15;

  v8f acc0 = {}; v8f acc1 = {};

  for (int kk = 0; kk < DMODEL; kk += KC) {
    #pragma unroll
    for (int e = 0; e < 8; ++e) {
      int idx = tid * 8 + e;
      int r = idx >> 5, c = idx & 31;
      At[r][c] = (_Float16)x[(size_t)(s0 + r) * DMODEL + kk + c];
      Wt[r][c] = (_Float16)w[(size_t)(n0 + r) * DMODEL + kk + c];
    }
    if (kk + KC < DMODEL) {   // pull next chunk toward L2/L0 (global_prefetch_b8)
      __builtin_prefetch(&x[(size_t)(s0 + (tid >> 2)) * DMODEL + kk + KC], 0, 0);
      __builtin_prefetch(&w[(size_t)(n0 + (tid >> 2)) * DMODEL + kk + KC], 0, 0);
    }
    __syncthreads();

    v16h a;
    const int arow = tm * 16 + nlane;
    #pragma unroll
    for (int h = 0; h < 16; ++h) a[h] = At[arow][a_kmap(h, lane)];

    v16h b0, b1;
    const int kb = (lane >> 4) * 16;
    #pragma unroll
    for (int h = 0; h < 16; ++h) {
      b0[h] = Wt[tn0 * 16 + nlane][kb + h];
      b1[h] = Wt[(tn0 + 1) * 16 + nlane][kb + h];
    }
    acc0 = __builtin_amdgcn_wmma_f32_16x16x32_f16(false, a, false, b0, (short)0, acc0, false, false);
    acc1 = __builtin_amdgcn_wmma_f32_16x16x32_f16(false, a, false, b1, (short)0, acc1, false, false);
    __syncthreads();
  }

  // Epilogue: RoPE (for q,k) pairs adjacent head dims => adjacent lanes.
  #pragma unroll
  for (int t = 0; t < 2; ++t) {
    v8f acc = (t == 0) ? acc0 : acc1;
    const int tn = tn0 + t;
    const int d  = tn * 16 + nlane;                 // 0..63 within head
    const float invf = __powf(10000.0f, -(float)(d >> 1) * (1.0f / 32.0f));
    const bool even = (d & 1) == 0;
    #pragma unroll
    for (int r = 0; r < 8; ++r) {
      const int srow = s0 + tm * 16 + mbase + r;
      float val = acc[r];
      if (which < 2) {                              // uniform branch per block
        float ang = (float)pos[srow] * invf;
        float sn, cs;
        __sincosf(ang, &sn, &cs);
        float partner = __shfl_xor(val, 1, 32);
        val = even ? (val * cs - partner * sn) : (partner * sn + val * cs);
      }
      dst[((size_t)head * S_LEN + srow) * DK + d] = (_Float16)val;
    }
  }
}

// ---------------------------------------------------------------------------
// Pass 2: causal flash attention per head. grid = (S/64, heads), block = 256.
// K/V/Q tiles fetched by the Tensor Data Mover (wave 0 issues, TENSORcnt
// waits, workgroup barrier releases consumers).
// ---------------------------------------------------------------------------
__global__ __launch_bounds__(256)
void flash_attn_kernel(const _Float16* __restrict__ qh,
                       const _Float16* __restrict__ kh,
                       const _Float16* __restrict__ vh,
                       _Float16* __restrict__ att)
{
  const int i0   = blockIdx.x * TILE;
  const int head = blockIdx.y;
  const int tid  = threadIdx.x;
  const int wave = tid >> 5;
  const int lane = tid & 31;
  const int tm    = wave >> 1;
  const int tn0   = (wave & 1) * 2;
  const int mbase = (lane >> 4) * 8;
  const int nlane = lane & 15;

  __shared__ _Float16 Qs[TILE][DK + 8];
  __shared__ _Float16 Ks[TILE][DK + 8];
  __shared__ _Float16 Vs[TILE][DK + 8];
  __shared__ _Float16 Ps[TILE][DK + 8];
  __shared__ float    Ss[TILE][TILE + 4];
  __shared__ float    mrow[TILE], lrow[TILE], corr[TILE];

  // ---- load Q tile once ----
#if HAVE_TDM
  if (wave == 0) {
    tdm_load_tile((unsigned int)(size_t)&Qs[0][0],
                  qh + ((size_t)head * S_LEN + i0) * DK);
  }
#else
  #pragma unroll
  for (int e = 0; e < 16; ++e) {
    int idx = tid * 16 + e;
    int r = idx >> 6, c = idx & 63;
    Qs[r][c] = qh[((size_t)head * S_LEN + i0 + r) * DK + c];
  }
#endif
  if (tid < TILE) { mrow[tid] = -3.0e38f; lrow[tid] = 0.0f; }

  v8f o0 = {}; v8f o1 = {};

  for (int j0 = 0; j0 <= i0; j0 += TILE) {
    // ---- stage K and V tiles ----
#if HAVE_TDM
    if (wave == 0) {
      tdm_load_tile((unsigned int)(size_t)&Ks[0][0],
                    kh + ((size_t)head * S_LEN + j0) * DK);
      tdm_load_tile((unsigned int)(size_t)&Vs[0][0],
                    vh + ((size_t)head * S_LEN + j0) * DK);
      __builtin_amdgcn_s_wait_tensorcnt(0);   // covers Q on first iteration too
    }
#else
    #pragma unroll
    for (int e = 0; e < 16; ++e) {
      int idx = tid * 16 + e;
      int r = idx >> 6, c = idx & 63;
      Ks[r][c] = kh[((size_t)head * S_LEN + j0 + r) * DK + c];
      Vs[r][c] = vh[((size_t)head * S_LEN + j0 + r) * DK + c];
    }
#endif
    __syncthreads();

    // ---- S = (Q K^T) * 1/sqrt(64), both operands d-major in LDS ----
    v8f sa0 = {}; v8f sa1 = {};
    #pragma unroll
    for (int kk = 0; kk < DK; kk += KC) {
      v16h a;
      const int arow = tm * 16 + nlane;
      #pragma unroll
      for (int h = 0; h < 16; ++h) a[h] = Qs[arow][kk + a_kmap(h, lane)];
      v16h b0, b1;
      const int kb = (lane >> 4) * 16;
      #pragma unroll
      for (int h = 0; h < 16; ++h) {
        b0[h] = Ks[tn0 * 16 + nlane][kk + kb + h];
        b1[h] = Ks[(tn0 + 1) * 16 + nlane][kk + kb + h];
      }
      sa0 = __builtin_amdgcn_wmma_f32_16x16x32_f16(false, a, false, b0, (short)0, sa0, false, false);
      sa1 = __builtin_amdgcn_wmma_f32_16x16x32_f16(false, a, false, b1, (short)0, sa1, false, false);
    }
    const bool diag = (j0 == i0);   // only the diagonal tile needs the mask
    #pragma unroll
    for (int r = 0; r < 8; ++r) {
      const int iloc  = tm * 16 + mbase + r;
      const int j0loc = tn0 * 16 + nlane;
      const int j1loc = (tn0 + 1) * 16 + nlane;
      float v0 = sa0[r] * 0.125f;
      float v1 = sa1[r] * 0.125f;
      if (diag && j0loc > iloc) v0 = -3.0e38f;
      if (diag && j1loc > iloc) v1 = -3.0e38f;
      Ss[iloc][j0loc] = v0;
      Ss[iloc][j1loc] = v1;
    }
    __syncthreads();

    // ---- online softmax: waves 0-3 own 16 rows each, lane pairs split cols
    if (wave < 4) {
      const int row = wave * 16 + nlane;
      const int cb  = (lane >> 4) * 32;
      float mx = -3.0e38f;
      #pragma unroll
      for (int c = 0; c < 32; ++c) mx = fmaxf(mx, Ss[row][cb + c]);
      mx = fmaxf(mx, __shfl_xor(mx, 16, 32));
      const float mold = mrow[row];
      const float mnew = fmaxf(mold, mx);
      float sum = 0.0f;
      #pragma unroll
      for (int c = 0; c < 32; ++c) {
        float p = __expf(Ss[row][cb + c] - mnew);
        Ps[row][cb + c] = (_Float16)p;
        sum += p;
      }
      sum += __shfl_xor(sum, 16, 32);
      if (lane < 16) {
        float cf = __expf(mold - mnew);
        corr[row] = cf;
        lrow[row] = lrow[row] * cf + sum;
        mrow[row] = mnew;
      }
    }
    __syncthreads();

    // ---- O = O*corr + P @ V ----
    #pragma unroll
    for (int r = 0; r < 8; ++r) {
      const float cf = corr[tm * 16 + mbase + r];
      o0[r] *= cf;
      o1[r] *= cf;
    }
    #pragma unroll
    for (int kk = 0; kk < TILE; kk += KC) {
      v16h a;
      const int arow = tm * 16 + nlane;
      #pragma unroll
      for (int h = 0; h < 16; ++h) a[h] = Ps[arow][kk + a_kmap(h, lane)];
      v16h b0, b1;
      const int kb = kk + (lane >> 4) * 16;
      #pragma unroll
      for (int h = 0; h < 16; ++h) {
        b0[h] = Vs[kb + h][tn0 * 16 + nlane];
        b1[h] = Vs[kb + h][(tn0 + 1) * 16 + nlane];
      }
      o0 = __builtin_amdgcn_wmma_f32_16x16x32_f16(false, a, false, b0, (short)0, o0, false, false);
      o1 = __builtin_amdgcn_wmma_f32_16x16x32_f16(false, a, false, b1, (short)0, o1, false, false);
    }
    __syncthreads();
  }

  #pragma unroll
  for (int r = 0; r < 8; ++r) {
    const int iloc = tm * 16 + mbase + r;
    const float inv_l = 1.0f / lrow[iloc];
    const size_t srow = (size_t)(i0 + iloc);
    att[srow * DMODEL + head * DK + tn0 * 16 + nlane]       = (_Float16)(o0[r] * inv_l);
    att[srow * DMODEL + head * DK + (tn0 + 1) * 16 + nlane] = (_Float16)(o1[r] * inv_l);
  }
}

// ---------------------------------------------------------------------------
// Pass 3: out = attn @ wo.T, fp32 output. grid = (S/64, D/64), block = 256.
// ---------------------------------------------------------------------------
__global__ __launch_bounds__(256)
void out_proj_kernel(const _Float16* __restrict__ att,
                     const float* __restrict__ wo,
                     float* __restrict__ out)
{
  const int s0 = blockIdx.x * TILE;
  const int n0 = blockIdx.y * TILE;

  __shared__ _Float16 At[TILE][KC + 8];
  __shared__ _Float16 Wt[TILE][KC + 8];

  const int tid   = threadIdx.x;
  const int wave  = tid >> 5;
  const int lane  = tid & 31;
  const int tm    = wave >> 1;
  const int tn0   = (wave & 1) * 2;
  const int mbase = (lane >> 4) * 8;
  const int nlane = lane & 15;

  v8f acc0 = {}; v8f acc1 = {};

  for (int kk = 0; kk < DMODEL; kk += KC) {
    #pragma unroll
    for (int e = 0; e < 8; ++e) {
      int idx = tid * 8 + e;
      int r = idx >> 5, c = idx & 31;
      At[r][c] = att[(size_t)(s0 + r) * DMODEL + kk + c];
      Wt[r][c] = (_Float16)wo[(size_t)(n0 + r) * DMODEL + kk + c];
    }
    if (kk + KC < DMODEL) {
      __builtin_prefetch(&att[(size_t)(s0 + (tid >> 2)) * DMODEL + kk + KC], 0, 0);
      __builtin_prefetch(&wo[(size_t)(n0 + (tid >> 2)) * DMODEL + kk + KC], 0, 0);
    }
    __syncthreads();

    v16h a;
    const int arow = tm * 16 + nlane;
    #pragma unroll
    for (int h = 0; h < 16; ++h) a[h] = At[arow][a_kmap(h, lane)];
    v16h b0, b1;
    const int kb = (lane >> 4) * 16;
    #pragma unroll
    for (int h = 0; h < 16; ++h) {
      b0[h] = Wt[tn0 * 16 + nlane][kb + h];
      b1[h] = Wt[(tn0 + 1) * 16 + nlane][kb + h];
    }
    acc0 = __builtin_amdgcn_wmma_f32_16x16x32_f16(false, a, false, b0, (short)0, acc0, false, false);
    acc1 = __builtin_amdgcn_wmma_f32_16x16x32_f16(false, a, false, b1, (short)0, acc1, false, false);
    __syncthreads();
  }

  #pragma unroll
  for (int r = 0; r < 8; ++r) {
    const int row = s0 + tm * 16 + mbase + r;
    out[(size_t)row * DMODEL + n0 + tn0 * 16 + nlane]       = acc0[r];
    out[(size_t)row * DMODEL + n0 + (tn0 + 1) * 16 + nlane] = acc1[r];
  }
}

// ---------------------------------------------------------------------------
extern "C" void kernel_launch(void* const* d_in, const int* in_sizes, int n_in,
                              void* d_out, int out_size, void* d_ws, size_t ws_size,
                              hipStream_t stream) {
  const float* x   = (const float*)d_in[0];
  const int*   pos = (const int*)d_in[1];
  const float* wq  = (const float*)d_in[2];
  const float* wk  = (const float*)d_in[3];
  const float* wv  = (const float*)d_in[4];
  const float* wo  = (const float*)d_in[5];
  float* out = (float*)d_out;

  const size_t per = (size_t)NHEADS * S_LEN * DK;   // f16 elements per tensor
  _Float16* qh  = (_Float16*)d_ws;
  _Float16* kh  = qh + per;
  _Float16* vh  = kh + per;
  _Float16* att = vh + per;                          // [S][DMODEL] f16

  qkv_rope_kernel<<<dim3(S_LEN / TILE, NHEADS, 3), 256, 0, stream>>>(
      x, wq, wk, wv, pos, qh, kh, vh);
  flash_attn_kernel<<<dim3(S_LEN / TILE, NHEADS), 256, 0, stream>>>(
      qh, kh, vh, att);
  out_proj_kernel<<<dim3(S_LEN / TILE, DMODEL / TILE), 256, 0, stream>>>(
      att, wo, out);
}
